// mimoGIN_20040317403499
// MI455X (gfx1250) — matrix operand order repaired and checked by reading, not compiled, not run on hardware
//
#include <hip/hip_runtime.h>
#include <hip/hip_bf16.h>

// ---------------- problem constants (match reference) ----------------
#define N_NODES   50000
#define N_EDGES   800000
#define N_GRAPHS  128
#define DIM       128      // IN == H == 128
#define N_CLASSES 10
#define N_ENS     2
#define N_LAYERS  4
#define BN_EPS    1e-5f

typedef __bf16 bf16;
typedef __bf16 v16bf __attribute__((ext_vector_type(16)));
typedef __bf16 v8bf  __attribute__((ext_vector_type(8)));
typedef float  v8f   __attribute__((ext_vector_type(8)));
typedef int    v4i   __attribute__((ext_vector_type(4)));

typedef __attribute__((address_space(1))) v4i* gptr_b128;   // global b128 chunk
typedef __attribute__((address_space(3))) v4i* lptr_b128;   // LDS b128 chunk

// ---------------- weight swizzle: row-major f32 [K=128][N=128] ->
// WMMA-B fragment-major bf16.  Tile t=(kt*8+nt), lane l, elem j:
//   col n = nt*16 + (l&15), k = kt*32 + (l>>4)*16 + j
// so each lane's 16 operands are contiguous (32B) in the swizzled buffer.
__global__ void k_swizzle_w(const float* __restrict__ src, bf16* __restrict__ dst) {
    int tid = blockIdx.x * blockDim.x + threadIdx.x;
    if (tid >= DIM * DIM) return;
    int t    = tid >> 9;          // 0..31 tiles
    int r    = tid & 511;
    int lane = r >> 4;
    int j    = r & 15;
    int kt = t >> 3, nt = t & 7;
    int n = nt * 16 + (lane & 15);
    int k = kt * 32 + (lane >> 4) * 16 + j;
    dst[tid] = (bf16)src[k * DIM + n];
}

__global__ void k_f32_to_bf16(const float* __restrict__ src, bf16* __restrict__ dst, int n) {
    int tid = blockIdx.x * blockDim.x + threadIdx.x;
    if (tid < n) dst[tid] = (bf16)src[tid];
}

__global__ void k_zero_f32(float* __restrict__ p, int n) {
    int tid = blockIdx.x * blockDim.x + threadIdx.x;
    if (tid < n) p[tid] = 0.0f;
}

// scatter-add: agg[dst[e]] += x[src[e]] ; one thread handles 4 features of one edge
__global__ void k_scatter_add(const bf16* __restrict__ x,
                              const int* __restrict__ esrc,
                              const int* __restrict__ edst,
                              float* __restrict__ agg) {
    int tid = blockIdx.x * blockDim.x + threadIdx.x;
    if (tid >= N_EDGES * 32) return;
    int e = tid >> 5;
    int c = (tid & 31) * 4;
    int s = esrc[e], d = edst[e];
    const bf16* xp = x + (long)s * DIM + c;
    float* ap = agg + (long)d * DIM + c;
#pragma unroll
    for (int q = 0; q < 4; ++q)
        atomicAdd(ap + q, (float)xp[q]);
}

// a_in = bf16(agg + (1+eps)*x)
__global__ void k_combine(const float* __restrict__ agg, const bf16* __restrict__ x,
                          const float* __restrict__ eps_ptr, int eps_idx,
                          bf16* __restrict__ out) {
    int tid = blockIdx.x * blockDim.x + threadIdx.x;
    if (tid >= N_NODES * DIM) return;
    float e = 1.0f + eps_ptr[eps_idx];
    out[tid] = (bf16)(agg[tid] + e * (float)x[tid]);
}

// Fused GEMM: out = act(A[N x 128] @ W[128 x 128] + b), bf16 in/out, f32 WMMA accum.
// Weight matrix (32KB, fragment-major) is staged once per workgroup into LDS via
// the CDNA5 async global->LDS path (ASYNCcnt), then each wave computes
// 16 rows x 128 cols: 4 K-steps x 8 N-tiles of v_wmma_f32_16x16x32_bf16 with B
// fragments from ds_load_b128.
// do_bn==0: relu(v+b)           (first MLP half)
// do_bn==1: relu(BN(relu(v+b))) (second half + outer layer relu)
__global__ __launch_bounds__(256)
void k_gemm_mlp(const bf16* __restrict__ A, const bf16* __restrict__ Wswz,
                const float* __restrict__ bias,
                const float* __restrict__ gamma, const float* __restrict__ beta,
                const float* __restrict__ mean,  const float* __restrict__ var,
                int do_bn, bf16* __restrict__ out) {
    __shared__ bf16 wlds[DIM * DIM];           // 32 KB
    const int tid  = threadIdx.x;
    const int lane = threadIdx.x & 31;
    const int wave = threadIdx.x >> 5;

    // ---- cooperative stage of W into LDS: 2048 x 16B chunks over 256 threads ----
#if __has_builtin(__builtin_amdgcn_global_load_async_to_lds_b128)
#pragma unroll
    for (int i = 0; i < 8; ++i) {
        int c = i * 256 + tid;                 // chunk index, 16 bytes each
        __builtin_amdgcn_global_load_async_to_lds_b128(
            (gptr_b128)((const char*)Wswz + (size_t)c * 16),
            (lptr_b128)((char*)&wlds[0] + (size_t)c * 16),
            0, 0);
    }
    asm volatile("s_wait_asynccnt 0x0" ::: "memory");
#else
#pragma unroll
    for (int i = 0; i < 8; ++i) {
        int c = i * 256 + tid;
        *(v8bf*)((char*)&wlds[0] + (size_t)c * 16) =
            *(const v8bf*)((const char*)Wswz + (size_t)c * 16);
    }
#endif
    __syncthreads();

    const int rowbase = (blockIdx.x * 8 + wave) * 16;
    const int half = lane >> 4;

    int arow = rowbase + (lane & 15);
    if (arow >= N_NODES) arow = N_NODES - 1;   // clamp loads; stores are guarded
    const bf16* Arow = A + (long)arow * DIM;

    v8f acc[8];
#pragma unroll
    for (int nt = 0; nt < 8; ++nt)
#pragma unroll
        for (int r = 0; r < 8; ++r) acc[nt][r] = 0.0f;

#pragma unroll
    for (int kt = 0; kt < 4; ++kt) {
        // A fragment (16-bit A 16x32 layout): lane holds row M=lane&15,
        // K = kt*32 + {half*16 .. half*16+15} (contiguous, matching B swizzle)
        v8bf lo = *(const v8bf*)(Arow + kt * 32 + half * 16);
        v8bf hi = *(const v8bf*)(Arow + kt * 32 + half * 16 + 8);
        v16bf a;
#pragma unroll
        for (int j = 0; j < 8; ++j) { a[j] = lo[j]; a[8 + j] = hi[j]; }
#pragma unroll
        for (int nt = 0; nt < 8; ++nt) {
            v16bf b = *(const v16bf*)(&wlds[(kt * 8 + nt) * 512 + lane * 16]);
            acc[nt] = __builtin_amdgcn_wmma_f32_16x16x32_bf16(
                false, a, false, b, (short)0, acc[nt], false, false);
        }
    }

    // D layout: col = lane&15 (within N-tile), row = rowbase + half*8 + r
    const int ncol0 = lane & 15;
    const int rbase = rowbase + half * 8;
#pragma unroll
    for (int nt = 0; nt < 8; ++nt) {
        int col = nt * 16 + ncol0;
        float bia = bias[col];
        float g = 0.f, bt = 0.f, mn = 0.f, rs = 0.f;
        if (do_bn) {
            g = gamma[col]; bt = beta[col]; mn = mean[col];
            rs = rsqrtf(var[col] + BN_EPS);
        }
#pragma unroll
        for (int r = 0; r < 8; ++r) {
            int row = rbase + r;
            if (row < N_NODES) {
                float v = acc[nt][r] + bia;
                v = v > 0.f ? v : 0.f;                 // relu(h@W + b)
                if (do_bn) {
                    v = g * (v - mn) * rs + bt;        // BatchNorm (eval)
                    v = v > 0.f ? v : 0.f;             // outer layer relu
                }
                out[(long)row * DIM + col] = (bf16)v;
            }
        }
    }
}

__global__ void k_pool_cnt(const int* __restrict__ batch, float* __restrict__ cnt) {
    int n = blockIdx.x * blockDim.x + threadIdx.x;
    if (n < N_NODES) atomicAdd(&cnt[batch[n]], 1.0f);
}

__global__ void k_pool_feat(const bf16* __restrict__ h, const int* __restrict__ batch,
                            float* __restrict__ pooled) {
    int tid = blockIdx.x * blockDim.x + threadIdx.x;
    if (tid >= N_NODES * 32) return;
    int n = tid >> 5;
    int c = (tid & 31) * 4;
    int b = batch[n];
    const bf16* hp = h + (long)n * DIM + c;
    float* pp = pooled + (long)b * DIM + c;
#pragma unroll
    for (int q = 0; q < 4; ++q)
        atomicAdd(pp + q, (float)hp[q]);
}

// per-graph head: mean-pool normalize + lin1/relu + lin2   (tiny: scalar f32)
__global__ __launch_bounds__(128)
void k_head(const float* __restrict__ pooled, const float* __restrict__ cnt,
            const float* __restrict__ W1, const float* __restrict__ b1,
            const float* __restrict__ W2, const float* __restrict__ b2,
            float* __restrict__ out) {
    __shared__ float pg[DIM];
    __shared__ float z[DIM];
    int g = blockIdx.x;
    int t = threadIdx.x;
    float inv = 1.0f / fmaxf(cnt[g], 1.0f);
    pg[t] = pooled[(long)g * DIM + t] * inv;
    __syncthreads();
    float s = b1[t];
    for (int k = 0; k < DIM; ++k) s += pg[k] * W1[k * DIM + t];
    z[t] = fmaxf(s, 0.0f);
    __syncthreads();
    if (t < N_CLASSES) {
        float y = b2[t];
        for (int k = 0; k < DIM; ++k) y += z[k] * W2[k * N_CLASSES + t];
        out[(long)g * N_CLASSES + t] = y;
    }
}

// ---------------------------------------------------------------------
static inline size_t align256(size_t x) { return (x + 255) & ~(size_t)255; }

extern "C" void kernel_launch(void* const* d_in, const int* in_sizes, int n_in,
                              void* d_out, int out_size, void* d_ws, size_t ws_size,
                              hipStream_t stream) {
    const float* x     = (const float*)d_in[0];
    const int*   edge  = (const int*)d_in[1];
    const int*   batch = (const int*)d_in[2];
    const float* c1W1  = (const float*)d_in[3];
    const float* c1b1  = (const float*)d_in[4];
    const float* c1W2  = (const float*)d_in[5];
    const float* c1b2  = (const float*)d_in[6];
    const float* c1g   = (const float*)d_in[7];
    const float* c1be  = (const float*)d_in[8];
    const float* c1m   = (const float*)d_in[9];
    const float* c1v   = (const float*)d_in[10];
    const float* c1eps = (const float*)d_in[11];
    const float* csW1  = (const float*)d_in[12];
    const float* csb1  = (const float*)d_in[13];
    const float* csW2  = (const float*)d_in[14];
    const float* csb2  = (const float*)d_in[15];
    const float* csg   = (const float*)d_in[16];
    const float* csbe  = (const float*)d_in[17];
    const float* csm   = (const float*)d_in[18];
    const float* csv   = (const float*)d_in[19];
    const float* cseps = (const float*)d_in[20];
    const float* l1W   = (const float*)d_in[21];
    const float* l1b   = (const float*)d_in[22];
    const float* l2W   = (const float*)d_in[23];
    const float* l2b   = (const float*)d_in[24];
    float* out = (float*)d_out;

    // ---- workspace carve ----
    char* w = (char*)d_ws;
    size_t off = 0;
    const size_t MAT = (size_t)DIM * DIM;            // 16384 elems
    bf16* swz = (bf16*)(w + off);  off = align256(off + 10 * MAT * sizeof(bf16));
    float* agg = (float*)(w + off); off = align256(off + (size_t)N_NODES * DIM * sizeof(float));
    bf16* xb0 = (bf16*)(w + off);  off = align256(off + (size_t)N_NODES * DIM * sizeof(bf16));
    bf16* xb1 = (bf16*)(w + off);  off = align256(off + (size_t)N_NODES * DIM * sizeof(bf16));
    bf16* a_in = (bf16*)(w + off); off = align256(off + (size_t)N_NODES * DIM * sizeof(bf16));
    bf16* h1  = (bf16*)(w + off);  off = align256(off + (size_t)N_NODES * DIM * sizeof(bf16));
    float* pooled = (float*)(w + off); off = align256(off + (size_t)N_GRAPHS * DIM * sizeof(float));
    float* cnt = (float*)(w + off);    off = align256(off + (size_t)N_GRAPHS * sizeof(float));
    (void)ws_size; (void)n_in; (void)in_sizes; (void)out_size;

    const int BLK = 256;
    const int swzGrid  = (DIM * DIM + BLK - 1) / BLK;              // 64
    const int elemGrid = (N_NODES * DIM + BLK - 1) / BLK;          // 25000
    const int scatGrid = (N_EDGES * 32 + BLK - 1) / BLK;           // 100000
    const int gemmGrid = ((N_NODES + 15) / 16 + 7) / 8;            // 391
    const int cntGrid  = (N_NODES + BLK - 1) / BLK;
    const int poolGrid = (N_NODES * 32 + BLK - 1) / BLK;

    // ---- pre-swizzle the 10 weight matrices to WMMA-B fragment layout ----
    // slots: 0..3 = conv1 W1/W2 for ensembles 0,1 ; 4+2l / 5+2l = convs layer l W1/W2
    for (int e = 0; e < N_ENS; ++e) {
        k_swizzle_w<<<swzGrid, BLK, 0, stream>>>(c1W1 + (size_t)e * MAT, swz + (size_t)(e * 2 + 0) * MAT);
        k_swizzle_w<<<swzGrid, BLK, 0, stream>>>(c1W2 + (size_t)e * MAT, swz + (size_t)(e * 2 + 1) * MAT);
    }
    for (int l = 0; l < N_LAYERS - 1; ++l) {
        k_swizzle_w<<<swzGrid, BLK, 0, stream>>>(csW1 + (size_t)l * MAT, swz + (size_t)(4 + 2 * l) * MAT);
        k_swizzle_w<<<swzGrid, BLK, 0, stream>>>(csW2 + (size_t)l * MAT, swz + (size_t)(5 + 2 * l) * MAT);
    }

    for (int e = 0; e < N_ENS; ++e) {
        const int* esrc = edge + (size_t)e * 2 * N_EDGES;
        const int* edst = esrc + N_EDGES;
        const int* bat  = batch + (size_t)e * N_NODES;

        bf16* xb[2] = {xb0, xb1};
        int cur = 0;
        k_f32_to_bf16<<<elemGrid, BLK, 0, stream>>>(x + (size_t)e * N_NODES * DIM, xb[cur], N_NODES * DIM);

        for (int l = 0; l < N_LAYERS; ++l) {
            k_zero_f32<<<elemGrid, BLK, 0, stream>>>(agg, N_NODES * DIM);
            k_scatter_add<<<scatGrid, BLK, 0, stream>>>(xb[cur], esrc, edst, agg);

            const float *epsP, *b1p, *b2p, *gp, *bep, *mp, *vp;
            const bf16 *w1s, *w2s;
            int epsI;
            if (l == 0) {
                epsP = c1eps; epsI = e;
                w1s = swz + (size_t)(e * 2 + 0) * MAT;  w2s = swz + (size_t)(e * 2 + 1) * MAT;
                b1p = c1b1 + (size_t)e * DIM;  b2p = c1b2 + (size_t)e * DIM;
                gp = c1g + (size_t)e * DIM; bep = c1be + (size_t)e * DIM;
                mp = c1m + (size_t)e * DIM; vp  = c1v  + (size_t)e * DIM;
            } else {
                int s = l - 1;
                epsP = cseps; epsI = s;
                w1s = swz + (size_t)(4 + 2 * s) * MAT;  w2s = swz + (size_t)(5 + 2 * s) * MAT;
                b1p = csb1 + (size_t)s * DIM;  b2p = csb2 + (size_t)s * DIM;
                gp = csg + (size_t)s * DIM; bep = csbe + (size_t)s * DIM;
                mp = csm + (size_t)s * DIM; vp  = csv  + (size_t)s * DIM;
            }

            k_combine<<<elemGrid, BLK, 0, stream>>>(agg, xb[cur], epsP, epsI, a_in);
            k_gemm_mlp<<<gemmGrid, BLK, 0, stream>>>(a_in, w1s, b1p, gp, bep, mp, vp, 0, h1);
            k_gemm_mlp<<<gemmGrid, BLK, 0, stream>>>(h1,  w2s, b2p, gp, bep, mp, vp, 1, xb[1 - cur]);
            cur ^= 1;
        }

        k_zero_f32<<<(N_GRAPHS * DIM + BLK - 1) / BLK, BLK, 0, stream>>>(pooled, N_GRAPHS * DIM);
        k_zero_f32<<<1, BLK, 0, stream>>>(cnt, N_GRAPHS);
        k_pool_cnt<<<cntGrid, BLK, 0, stream>>>(bat, cnt);
        k_pool_feat<<<poolGrid, BLK, 0, stream>>>(xb[cur], bat, pooled);
        k_head<<<N_GRAPHS, 128, 0, stream>>>(pooled, cnt,
                                             l1W + (size_t)e * MAT, l1b + (size_t)e * DIM,
                                             l2W + (size_t)e * DIM * N_CLASSES, l2b + (size_t)e * N_CLASSES,
                                             out + (size_t)e * N_GRAPHS * N_CLASSES);
    }
}
